// RelaPro_32547262169374
// MI455X (gfx1250) — matrix operand order repaired and checked by reading, not compiled, not run on hardware
//
#include <hip/hip_runtime.h>
#include <hip/hip_bf16.h>
#include <math.h>

typedef __attribute__((ext_vector_type(16))) __bf16 v16bf;
typedef __attribute__((ext_vector_type(4)))  __bf16 v4bf;
typedef __attribute__((ext_vector_type(8)))  float  v8f;

#define NROWS   131072
#define LDIM    300
#define KDIM    600            // 2*L
#define KPAD    608            // 19 k-tiles of 32
#define NPAD    304            // 19 col-tiles of 16
#define NT      19
#define KT      19
#define APITCH  616            // LDS pitch in bf16 elems (16B aligned, bank-skewed)
#define ROWS_PER_BLOCK 128
#define GEMM_BLOCK 256         // 8 waves * 32

// ---- workspace layout (bytes) ----
#define W1T_ELEMS   (NPAD * KPAD)                 // 184832 bf16
#define WS_LAN_OFF  (W1T_ELEMS * 2)               // 369664
#define WS_ACC_OFF  (WS_LAN_OFF + NROWS * 4)      // 893952 ; 8 floats of accum

// ------------------------------------------------------------------
// Kernel 0: W1 (600x300 f32, row-major) -> W1T (304x608 bf16, [n][k]),
// zero-padded; also zero the global accumulators.
// ------------------------------------------------------------------
__global__ void k_convert(const float* __restrict__ w1,
                          __bf16* __restrict__ w1t,
                          float* __restrict__ acc) {
    int idx = blockIdx.x * 256 + threadIdx.x;
    if (idx < 8) acc[idx] = 0.0f;
    if (idx >= NPAD * KPAD) return;
    int n = idx / KPAD;
    int k = idx - n * KPAD;
    float v = (n < LDIM && k < KDIM) ? w1[(size_t)k * LDIM + n] : 0.0f;
    w1t[(size_t)n * KPAD + k] = (__bf16)v;
}

// ------------------------------------------------------------------
// Kernel 1: fused  lan_scores = relu(relu([sbj|obj]) @ W1 + b1) @ w2
// bf16 WMMA, f32 accumulate. One wave per 16-row M-tile.
// ------------------------------------------------------------------
__global__ __launch_bounds__(GEMM_BLOCK) void k_gemm(
    const float* __restrict__ sbj, const float* __restrict__ obj,
    const __bf16* __restrict__ w1t, const float* __restrict__ b1,
    const float* __restrict__ w2, float* __restrict__ lanOut)
{
    __shared__ __attribute__((aligned(16))) __bf16 lA[ROWS_PER_BLOCK * APITCH];
    __shared__ __attribute__((aligned(16))) __bf16 lB[16 * APITCH];

    const int tid  = threadIdx.x;
    const int lane = tid & 31;
    const int wv   = tid >> 5;           // 0..7
    const int hi   = lane >> 4;          // half of wave
    const int ln   = lane & 15;
    const int row0 = blockIdx.x * ROWS_PER_BLOCK;

    // ---- stage A = relu(concat(sbj,obj)) as bf16 into LDS (vectorized) ----
    for (int i = tid; i < ROWS_PER_BLOCK * (LDIM / 4); i += GEMM_BLOCK) {
        int r  = i / (LDIM / 4);
        int c4 = i - r * (LDIM / 4);
        const float4 vs = *(const float4*)(sbj + (size_t)(row0 + r) * LDIM + c4 * 4);
        const float4 vo = *(const float4*)(obj + (size_t)(row0 + r) * LDIM + c4 * 4);
        v4bf bs, bo;
        bs[0] = (__bf16)fmaxf(vs.x, 0.f); bs[1] = (__bf16)fmaxf(vs.y, 0.f);
        bs[2] = (__bf16)fmaxf(vs.z, 0.f); bs[3] = (__bf16)fmaxf(vs.w, 0.f);
        bo[0] = (__bf16)fmaxf(vo.x, 0.f); bo[1] = (__bf16)fmaxf(vo.y, 0.f);
        bo[2] = (__bf16)fmaxf(vo.z, 0.f); bo[3] = (__bf16)fmaxf(vo.w, 0.f);
        *(v4bf*)&lA[r * APITCH + c4 * 4]        = bs;
        *(v4bf*)&lA[r * APITCH + LDIM + c4 * 4] = bo;
    }
    // zero K padding 600..607
    for (int i = tid; i < ROWS_PER_BLOCK * 2; i += GEMM_BLOCK) {
        v4bf z = {};
        *(v4bf*)&lA[(i >> 1) * APITCH + KDIM + (i & 1) * 4] = z;
    }

    float sacc[8] = {0.f, 0.f, 0.f, 0.f, 0.f, 0.f, 0.f, 0.f};
    const int arow = wv * 16 + ln;

    for (int nt = 0; nt < NT; ++nt) {
        __syncthreads();   // A ready / previous tile's compute done
        // ---- stage B tile (16 cols of W1T, K-contiguous) ----
        for (int i = tid; i < 16 * (KPAD / 8); i += GEMM_BLOCK) {
            int nn = i / (KPAD / 8);
            int k8 = i - nn * (KPAD / 8);
            *(float4*)&lB[nn * APITCH + k8 * 8] =
                *(const float4*)&w1t[(size_t)(nt * 16 + nn) * KPAD + k8 * 8];
        }
        __syncthreads();

        v8f c = {};
#pragma unroll
        for (int kt = 0; kt < KT; ++kt) {
            v16bf a, b;
            const int ak = kt * 32 + hi * 8;      // A: lane holds K = ak..ak+7, ak+16..ak+23
            *((float4*)&a)       = *(const float4*)&lA[arow * APITCH + ak];
            *(((float4*)&a) + 1) = *(const float4*)&lA[arow * APITCH + ak + 16];
            const int bk = kt * 32 + hi * 8;      // B: lane (col ln) holds K = bk..bk+7, bk+16..bk+23
            *((float4*)&b)       = *(const float4*)&lB[ln * APITCH + bk];
            *(((float4*)&b) + 1) = *(const float4*)&lB[ln * APITCH + bk + 16];
            c = __builtin_amdgcn_wmma_f32_16x16x32_bf16(
                    false, a, false, b, (short)0, c, false, false);
        }

        // fuse: h = relu(c + b1[n]);  sacc += h * w2[n]
        const int n = nt * 16 + ln;
        const float b1v = (n < LDIM) ? b1[n] : 0.0f;
        const float w2v = (n < LDIM) ? w2[n] : 0.0f;
#pragma unroll
        for (int r = 0; r < 8; ++r) {
            float h = c[r] + b1v;
            h = fmaxf(h, 0.0f);
            sacc[r] += h * w2v;
        }
    }

    // reduce over the 16 lanes of each half (N dimension of the C layout)
#pragma unroll
    for (int m = 1; m < 16; m <<= 1) {
#pragma unroll
        for (int r = 0; r < 8; ++r) sacc[r] += __shfl_xor(sacc[r], m);
    }
    if (ln == 0) {
        const int rowbase = row0 + wv * 16 + hi * 8;   // VGPR r -> row r (+8 for hi half)
#pragma unroll
        for (int r = 0; r < 8; ++r) lanOut[rowbase + r] = sacc[r];
    }
}

// ------------------------------------------------------------------
// Kernel 2: box features + sigmoid + per-row loss/recall contributions
// ------------------------------------------------------------------
__global__ __launch_bounds__(256) void k_post(
    const float* __restrict__ rlts, const float* __restrict__ spa_w,
    const float* __restrict__ spa_b, const float* __restrict__ lan_b2,
    const float* __restrict__ lanS, float* __restrict__ out,
    float* __restrict__ acc)
{
    const int row = blockIdx.x * 256 + threadIdx.x;
    const float* r = rlts + (size_t)row * 15;
    const float sx1 = r[5],  sy1 = r[6],  sx2 = r[7],  sy2 = r[8];
    const float ox1 = r[10], oy1 = r[11], ox2 = r[12], oy2 = r[13];
    const float sw = sx2 - sx1, sh = sy2 - sy1;
    const float ow = ox2 - ox1, oh = oy2 - oy1;

    float f0 = (sx1 - ox1) / sw, f1 = (sy1 - oy1) / sh;
    float f2 = logf(sw / ow),    f3 = logf(sh / oh);
    float f4 = (ox1 - sx1) / ow, f5 = (oy1 - sy1) / oh;
    float f6 = logf(ow / sw),    f7 = logf(oh / sh);

    float box = f0 * spa_w[0] + f1 * spa_w[1] + f2 * spa_w[2] + f3 * spa_w[3]
              + f4 * spa_w[4] + f5 * spa_w[5] + f6 * spa_w[6] + f7 * spa_w[7]
              + spa_b[0];
    float z = lanS[row] + lan_b2[0] + box;
    float s = 1.0f / (1.0f + expf(-z));
    out[row] = s;

    const float ys  = (r[4] > 0.0f) ? 1.0f : 0.0f;
    float ls  = fmaxf(logf(fminf(fmaxf(s,        1e-12f), 1.0f)), -100.0f);
    float l1s = fmaxf(logf(fminf(fmaxf(1.0f - s, 1e-12f), 1.0f)), -100.0f);
    float loss = -(ys * ls + (1.0f - ys) * l1s);
    float pos = ys, neg = 1.0f - ys;
    float pr = (neg > 0.0f && s >= 0.5f) ? 1.0f : 0.0f;  // N_pos_right
    float nr = (pos > 0.0f && s <  0.5f) ? 1.0f : 0.0f;  // N_neg_right

#pragma unroll
    for (int m = 16; m >= 1; m >>= 1) {
        loss += __shfl_xor(loss, m);
        pos  += __shfl_xor(pos,  m);
        neg  += __shfl_xor(neg,  m);
        pr   += __shfl_xor(pr,   m);
        nr   += __shfl_xor(nr,   m);
    }
    if ((threadIdx.x & 31) == 0) {
        atomicAdd(&acc[0], loss);
        atomicAdd(&acc[1], pos);
        atomicAdd(&acc[2], neg);
        atomicAdd(&acc[3], pr);
        atomicAdd(&acc[4], nr);
    }
}

// ------------------------------------------------------------------
// Kernel 3: finalize scalars
// ------------------------------------------------------------------
__global__ void k_final(const float* __restrict__ acc, float* __restrict__ out) {
    if (threadIdx.x == 0) {
        out[NROWS + 0] = acc[0] / (float)NROWS;
        out[NROWS + 1] = acc[3] / acc[1];
        out[NROWS + 2] = acc[4] / acc[2];
        out[NROWS + 3] = (acc[3] + acc[4]) / (acc[1] + acc[2]);
    }
}

extern "C" void kernel_launch(void* const* d_in, const int* in_sizes, int n_in,
                              void* d_out, int out_size, void* d_ws, size_t ws_size,
                              hipStream_t stream) {
    const float* sbj    = (const float*)d_in[0];
    const float* obj    = (const float*)d_in[1];
    const float* rlts   = (const float*)d_in[2];
    const float* spa_w  = (const float*)d_in[3];
    const float* spa_b  = (const float*)d_in[4];
    const float* lan_w1 = (const float*)d_in[5];
    const float* lan_b1 = (const float*)d_in[6];
    const float* lan_w2 = (const float*)d_in[7];
    const float* lan_b2 = (const float*)d_in[8];

    char* ws = (char*)d_ws;
    __bf16* w1t  = (__bf16*)ws;
    float*  lanS = (float*)(ws + WS_LAN_OFF);
    float*  acc  = (float*)(ws + WS_ACC_OFF);
    float*  out  = (float*)d_out;

    k_convert<<<(NPAD * KPAD + 255) / 256, 256, 0, stream>>>(lan_w1, w1t, acc);
    k_gemm<<<NROWS / ROWS_PER_BLOCK, GEMM_BLOCK, 0, stream>>>(
        sbj, obj, w1t, lan_b1, lan_w2, lanS);
    k_post<<<NROWS / 256, 256, 0, stream>>>(rlts, spa_w, spa_b, lan_b2, lanS, out, acc);
    k_final<<<1, 32, 0, stream>>>(acc, out);
}
